// HarmonicTransformerLayer_22625887715628
// MI455X (gfx1250) — compile-verified, matrix operand order checked by reading, not run on hardware
//
#include <hip/hip_runtime.h>
#include <hip/hip_bf16.h>

typedef __attribute__((ext_vector_type(16))) _Float16 v16h;
typedef __attribute__((ext_vector_type(8)))  _Float16 v8h;
typedef __attribute__((ext_vector_type(8)))  float    v8f;

#define B_  2
#define S_  2048
#define D_  1024
#define H_  16
#define F_  4096
#define M_  (B_ * S_)   // 4096 rows

// ---------------------------------------------------------------------------
// helpers
// ---------------------------------------------------------------------------
__device__ inline v16h cat8(v8h lo, v8h hi) {
  return __builtin_shufflevector(lo, hi, 0,1,2,3,4,5,6,7,8,9,10,11,12,13,14,15);
}

// 16x32 f16 fragment from LDS (k-contiguous rows). Lane l (l16=l&15, half=l>>4):
//   elements 0..7  -> k =      8*half + 0..7
//   elements 8..15 -> k = 16 + 8*half + 0..7
__device__ inline v16h load_frag(const _Float16* rowbase, int half) {
  v8h lo = *(const v8h*)(rowbase + 8 * half);
  v8h hi = *(const v8h*)(rowbase + 16 + 8 * half);
  return cat8(lo, hi);
}

__device__ inline v8f wmma16(v16h a, v16h b, v8f c) {
  return __builtin_amdgcn_wmma_f32_16x16x32_f16(
      false, a, false, b, (short)0, c, false, false);
}

// CDNA5 async global->LDS copy (ASYNCcnt-tracked, no VGPR data path).
__device__ inline void async_copy_b128(const void* gptr, void* lptr) {
  unsigned l = (unsigned)(unsigned long long)lptr;  // low 32 bits = LDS offset
  asm volatile("global_load_async_to_lds_b128 %0, %1, off"
               :: "v"(l), "v"(gptr) : "memory");
}
__device__ inline void async_wait0() {
  asm volatile("s_wait_asynccnt 0x0" ::: "memory");
}

__device__ inline float spectral_act(float x) {
  const float c = 0.7978845608028654f;              // sqrt(2/pi)
  float g = 0.5f * x * (1.0f + tanhf(c * (x + 0.044715f * x * x * x)));
  return g + 0.5f * sinf(x);                        // ALPHA=0.5, BETA=1.0
}

// ---------------------------------------------------------------------------
// f32 -> f16 conversion
// ---------------------------------------------------------------------------
__global__ void cvt_f32_f16(const float* __restrict__ src,
                            _Float16* __restrict__ dst, size_t n) {
  size_t i = (size_t)blockIdx.x * blockDim.x + threadIdx.x;
  size_t stride = (size_t)gridDim.x * blockDim.x;
  for (; i < n; i += stride) dst[i] = (_Float16)src[i];
}

// f32 [K,N] -> f16 transposed [N,K] (LDS-tiled, coalesced both sides)
__global__ __launch_bounds__(256) void cvt_transpose_f16(
    const float* __restrict__ src, _Float16* __restrict__ dst, int K, int N) {
  __shared__ float tile[32][33];
  const int kt = blockIdx.y * 32, nt = blockIdx.x * 32;
  const int tx = threadIdx.x, ty = threadIdx.y;  // block (32, 8)
#pragma unroll
  for (int r = ty; r < 32; r += 8)
    tile[r][tx] = src[(size_t)(kt + r) * N + nt + tx];
  __syncthreads();
#pragma unroll
  for (int r = ty; r < 32; r += 8)
    dst[(size_t)(nt + r) * K + kt + tx] = (_Float16)tile[tx][r];
}

// ---------------------------------------------------------------------------
// WMMA GEMM: C[M,N] = A[M,K](f16) @ W[K,N] + bias, with W pre-transposed
// as WT[N,K] (k-contiguous).  block = 256 (8 waves, 4x2 of 32x32),
// macro-tile 128x64, k-step 32, async global->LDS staging.
//   EPI 0: f16 out   EPI 1: f32 out   EPI 2: spectral->f16   EPI 3: f16 out^T
// ---------------------------------------------------------------------------
template <int EPI>
__global__ __launch_bounds__(256) void gemm_wmma(
    const _Float16* __restrict__ A, const _Float16* __restrict__ WT,
    const float* __restrict__ bias, _Float16* __restrict__ outH,
    float* __restrict__ outF, int M, int N, int K) {
  __shared__ _Float16 sA[128 * 40];  // [m][k], row stride 40
  __shared__ _Float16 sB[64 * 40];   // [n][k]

  const int t = threadIdx.x;
  const int lane = t & 31, wave = t >> 5;
  const int half = lane >> 4, l16 = lane & 15;
  const int m0 = blockIdx.y * 128, n0 = blockIdx.x * 64;
  const int wm = (wave >> 1) * 32, wn = (wave & 1) * 32;

  v8f acc[2][2] = {};

  const int ar = t >> 1, aseg = (t & 1) * 16;  // A: 16 f16 per thread
  const int br = t >> 2, bseg = (t & 3) * 8;   // B: 8 f16 per thread

  for (int kt = 0; kt < K; kt += 32) {
    __syncthreads();
    {
      const _Float16* ga = A + (size_t)(m0 + ar) * K + kt + aseg;
      async_copy_b128(ga,     &sA[ar * 40 + aseg]);
      async_copy_b128(ga + 8, &sA[ar * 40 + aseg + 8]);
      const _Float16* gb = WT + (size_t)(n0 + br) * K + kt + bseg;
      async_copy_b128(gb, &sB[br * 40 + bseg]);
    }
    if (kt + 32 < K) {  // global_prefetch_b8 for next tiles
      __builtin_prefetch(A + (size_t)(m0 + ar) * K + kt + 32 + aseg, 0, 0);
      __builtin_prefetch(WT + (size_t)(n0 + br) * K + kt + 32 + bseg, 0, 0);
    }
    async_wait0();
    __syncthreads();

    v16h af[2], bf[2];
#pragma unroll
    for (int i = 0; i < 2; ++i)
      af[i] = load_frag(&sA[(wm + 16 * i + l16) * 40], half);
#pragma unroll
    for (int j = 0; j < 2; ++j)
      bf[j] = load_frag(&sB[(wn + 16 * j + l16) * 40], half);
#pragma unroll
    for (int i = 0; i < 2; ++i)
#pragma unroll
      for (int j = 0; j < 2; ++j)
        acc[i][j] = wmma16(af[i], bf[j], acc[i][j]);
  }

#pragma unroll
  for (int i = 0; i < 2; ++i)
#pragma unroll
    for (int j = 0; j < 2; ++j) {
      const int nn = n0 + wn + 16 * j + l16;      // C layout: n = lane&15
      const float bval = bias[nn];
      if (EPI == 3) {
        // transposed f16 store: fixed n per lane, 8 consecutive m -> b128
        v8h pk;
#pragma unroll
        for (int jj = 0; jj < 8; ++jj)
          pk[jj] = (_Float16)(acc[i][j][jj] + bval);
        *(v8h*)(outH + (size_t)nn * M + (m0 + wm + 16 * i + 8 * half)) = pk;
      } else {
#pragma unroll
        for (int jj = 0; jj < 8; ++jj) {
          const int m = m0 + wm + 16 * i + jj + 8 * half;  // m = vgpr + 8*half
          float v = acc[i][j][jj] + bval;
          if (EPI == 2) v = spectral_act(v);
          size_t idx = (size_t)m * N + nn;
          if (EPI == 1) outF[idx] = v;
          else          outH[idx] = (_Float16)v;
        }
      }
    }
}

// ---------------------------------------------------------------------------
// Flash attention with harmonic scores.  V supplied pre-transposed:
// Vt[n, m] with n = h*64+hd, m = b*S+s  (keys contiguous).
// grid = (S/64, B*H); block = 128 (4 waves, wave = 16 query rows)
// ---------------------------------------------------------------------------
__global__ __launch_bounds__(128) void flash_attn(
    const _Float16* __restrict__ Q, const _Float16* __restrict__ Kc,
    const _Float16* __restrict__ Vt, const float* __restrict__ resonance,
    const float* __restrict__ phase, _Float16* __restrict__ Oh) {
  __shared__ _Float16 sK[32 * 64];     // [key][hd]
  __shared__ _Float16 sVt[64 * 32];    // [hd][key]
  __shared__ _Float16 sP[4][16 * 32];  // per-wave P tile [q][key]

  const int t = threadIdx.x;
  const int lane = t & 31, wave = t >> 5;
  const int half = lane >> 4, l16 = lane & 15;
  const int bh = blockIdx.y, b = bh / H_, h = bh % H_;
  const int q0 = blockIdx.x * 64 + wave * 16;
  const float r = resonance[h], ph = phase[h];
  const float scale = 0.125f;  // 1/sqrt(64)

  const _Float16* qbase = Q + ((size_t)(b * S_ + q0 + l16)) * D_ + h * 64;
  v16h qf0 = cat8(*(const v8h*)(qbase + 8 * half),
                  *(const v8h*)(qbase + 16 + 8 * half));
  v16h qf1 = cat8(*(const v8h*)(qbase + 32 + 8 * half),
                  *(const v8h*)(qbase + 48 + 8 * half));

  float st_m[8], st_l[8];
  v8f o[4] = {};
#pragma unroll
  for (int j = 0; j < 8; ++j) { st_m[j] = -3.0e38f; st_l[j] = 0.f; }

  const int kr = t >> 2, kseg = (t & 3) * 16;  // K loader: 16 f16 / thread
  const int vr = t >> 1, vseg = (t & 1) * 16;  // Vt loader: 16 f16 / thread
  for (int kv = 0; kv < S_; kv += 32) {
    __syncthreads();
    {
      const _Float16* kg = Kc + ((size_t)(b * S_ + kv + kr)) * D_ + h * 64 + kseg;
      async_copy_b128(kg,     &sK[kr * 64 + kseg]);
      async_copy_b128(kg + 8, &sK[kr * 64 + kseg + 8]);
      const _Float16* vg = Vt + ((size_t)(h * 64 + vr)) * M_ + b * S_ + kv + vseg;
      async_copy_b128(vg,     &sVt[vr * 32 + vseg]);
      async_copy_b128(vg + 8, &sVt[vr * 32 + vseg + 8]);
    }
    async_wait0();
    __syncthreads();

    // scores for 32 keys: two 16x16 tiles, hd=64 reduced as 32+32
    v8f s0 = {}, s1 = {};
    {
      const _Float16* k0 = &sK[l16 * 64];
      const _Float16* k1 = &sK[(16 + l16) * 64];
      v16h k00 = cat8(*(const v8h*)(k0 + 8 * half), *(const v8h*)(k0 + 16 + 8 * half));
      v16h k01 = cat8(*(const v8h*)(k0 + 32 + 8 * half), *(const v8h*)(k0 + 48 + 8 * half));
      v16h k10 = cat8(*(const v8h*)(k1 + 8 * half), *(const v8h*)(k1 + 16 + 8 * half));
      v16h k11 = cat8(*(const v8h*)(k1 + 32 + 8 * half), *(const v8h*)(k1 + 48 + 8 * half));
      s0 = wmma16(qf0, k00, s0); s0 = wmma16(qf1, k01, s0);
      s1 = wmma16(qf0, k10, s1); s1 = wmma16(qf1, k11, s1);
    }

    float corr[8];
#pragma unroll
    for (int j = 0; j < 8; ++j) {
      float a = s0[j] * scale;  a += r * sinf(a + ph);
      float c = s1[j] * scale;  c += r * sinf(c + ph);
      float mx = fmaxf(a, c);
#pragma unroll
      for (int off = 1; off < 16; off <<= 1)
        mx = fmaxf(mx, __shfl_xor(mx, off, 32));
      float nm = fmaxf(st_m[j], mx);
      float p0 = __expf(a - nm), p1 = __expf(c - nm);
      s0[j] = p0; s1[j] = p1;
      float sm = p0 + p1;
#pragma unroll
      for (int off = 1; off < 16; off <<= 1) sm += __shfl_xor(sm, off, 32);
      corr[j] = __expf(st_m[j] - nm);
      st_l[j] = st_l[j] * corr[j] + sm;
      st_m[j] = nm;
    }
#pragma unroll
    for (int tt = 0; tt < 4; ++tt)
#pragma unroll
      for (int j = 0; j < 8; ++j) o[tt][j] *= corr[j];

    // P (C-layout) -> LDS -> A-fragment layout
#pragma unroll
    for (int j = 0; j < 8; ++j) {
      int row = j + 8 * half;
      sP[wave][row * 32 + l16]      = (_Float16)s0[j];
      sP[wave][row * 32 + 16 + l16] = (_Float16)s1[j];
    }
    __syncthreads();
    v16h pf = cat8(*(const v8h*)(&sP[wave][l16 * 32 + 8 * half]),
                   *(const v8h*)(&sP[wave][l16 * 32 + 16 + 8 * half]));
#pragma unroll
    for (int tt = 0; tt < 4; ++tt) {
      const _Float16* vb = &sVt[(tt * 16 + l16) * 32];
      v16h vf = cat8(*(const v8h*)(vb + 8 * half),
                     *(const v8h*)(vb + 16 + 8 * half));
      o[tt] = wmma16(pf, vf, o[tt]);
    }
  }

#pragma unroll
  for (int j = 0; j < 8; ++j) {
    float inv = 1.0f / st_l[j];
    int qrow = q0 + j + 8 * half;
    _Float16* dst = Oh + ((size_t)(b * S_ + qrow)) * D_ + h * 64;
#pragma unroll
    for (int tt = 0; tt < 4; ++tt)
      dst[tt * 16 + l16] = (_Float16)(o[tt][j] * inv);
  }
}

// ---------------------------------------------------------------------------
// Harmonic LayerNorm (unbiased std): one block per row.
// ---------------------------------------------------------------------------
__global__ __launch_bounds__(256) void harmonic_ln(
    const float* __restrict__ A, const float* __restrict__ Bv,
    const float* __restrict__ g, const float* __restrict__ be,
    const float* __restrict__ s, float* __restrict__ outF,
    _Float16* __restrict__ outH) {
  __shared__ float red[256];
  const int row = blockIdx.x, t = threadIdx.x;
  const float* a = A + (size_t)row * D_;
  const float* bb = Bv + (size_t)row * D_;

  float sum = 0.f, sq = 0.f;
  for (int c = t; c < D_; c += 256) {
    float v = a[c] + bb[c];
    sum += v; sq += v * v;
  }
  red[t] = sum; __syncthreads();
  for (int off = 128; off > 0; off >>= 1) {
    if (t < off) red[t] += red[t + off];
    __syncthreads();
  }
  float tot = red[0]; __syncthreads();
  red[t] = sq; __syncthreads();
  for (int off = 128; off > 0; off >>= 1) {
    if (t < off) red[t] += red[t + off];
    __syncthreads();
  }
  float tot2 = red[0];

  float mean = tot / (float)D_;
  float var = (tot2 - mean * tot) / (float)(D_ - 1);
  float sd = sqrtf(fmaxf(var, 0.f));
  sd = sd + s[0] * sinf(sd * 3.14159265358979323846f);
  float inv = 1.0f / (sd + 1e-6f);

  for (int c = t; c < D_; c += 256) {
    float v = (a[c] + bb[c] - mean) * inv * g[c] + be[c];
    outF[(size_t)row * D_ + c] = v;
    if (outH) outH[(size_t)row * D_ + c] = (_Float16)v;
  }
}

// ---------------------------------------------------------------------------
// launch
// ---------------------------------------------------------------------------
extern "C" void kernel_launch(void* const* d_in, const int* in_sizes, int n_in,
                              void* d_out, int out_size, void* d_ws,
                              size_t ws_size, hipStream_t stream) {
  const float* x   = (const float*)d_in[0];
  const float* Wq  = (const float*)d_in[1];  const float* bq  = (const float*)d_in[2];
  const float* Wk  = (const float*)d_in[3];  const float* bk  = (const float*)d_in[4];
  const float* Wv  = (const float*)d_in[5];  const float* bv  = (const float*)d_in[6];
  const float* Wo  = (const float*)d_in[7];  const float* bo  = (const float*)d_in[8];
  const float* res = (const float*)d_in[9];  const float* phs = (const float*)d_in[10];
  const float* g1  = (const float*)d_in[11]; const float* b1  = (const float*)d_in[12];
  const float* s1  = (const float*)d_in[13];
  const float* W1  = (const float*)d_in[14]; const float* bf1 = (const float*)d_in[15];
  const float* W2  = (const float*)d_in[16]; const float* bf2 = (const float*)d_in[17];
  const float* g2  = (const float*)d_in[18]; const float* b2  = (const float*)d_in[19];
  const float* s2  = (const float*)d_in[20];

  char* ws = (char*)d_ws;
  size_t off = 0;
  auto alloc = [&](size_t bytes) {
    char* p = ws + off;
    off += (bytes + 255) & ~(size_t)255;
    return p;
  };
  _Float16* xb   = (_Float16*)alloc((size_t)M_ * D_ * 2);
  _Float16* wqb  = (_Float16*)alloc((size_t)D_ * D_ * 2);  // transposed [N,K]
  _Float16* wkb  = (_Float16*)alloc((size_t)D_ * D_ * 2);
  _Float16* wvb  = (_Float16*)alloc((size_t)D_ * D_ * 2);
  _Float16* wob  = (_Float16*)alloc((size_t)D_ * D_ * 2);
  _Float16* w1b  = (_Float16*)alloc((size_t)D_ * F_ * 2);
  _Float16* w2b  = (_Float16*)alloc((size_t)F_ * D_ * 2);
  _Float16* qb   = (_Float16*)alloc((size_t)M_ * D_ * 2);
  _Float16* kb2  = (_Float16*)alloc((size_t)M_ * D_ * 2);
  _Float16* vt   = (_Float16*)alloc((size_t)M_ * D_ * 2);  // V transposed [D,M]
  _Float16* at16 = (_Float16*)alloc((size_t)M_ * D_ * 2);
  float*    aprj = (float*)   alloc((size_t)M_ * D_ * 4);
  float*    y1   = (float*)   alloc((size_t)M_ * D_ * 4);
  _Float16* y1b  = (_Float16*)alloc((size_t)M_ * D_ * 2);
  _Float16* act  = (_Float16*)alloc((size_t)M_ * F_ * 2);
  float*    ffn  = (float*)   alloc((size_t)M_ * D_ * 4);

  const dim3 blk256(256), blk128(128), blkT(32, 8);

  cvt_f32_f16<<<2048, blk256, 0, stream>>>(x, xb, (size_t)M_ * D_);
  // weights: f32 [K,N] -> f16 transposed [N,K]
  dim3 gTdd(D_ / 32, D_ / 32);
  cvt_transpose_f16<<<gTdd, blkT, 0, stream>>>(Wq, wqb, D_, D_);
  cvt_transpose_f16<<<gTdd, blkT, 0, stream>>>(Wk, wkb, D_, D_);
  cvt_transpose_f16<<<gTdd, blkT, 0, stream>>>(Wv, wvb, D_, D_);
  cvt_transpose_f16<<<gTdd, blkT, 0, stream>>>(Wo, wob, D_, D_);
  cvt_transpose_f16<<<dim3(F_ / 32, D_ / 32), blkT, 0, stream>>>(W1, w1b, D_, F_);
  cvt_transpose_f16<<<dim3(D_ / 32, F_ / 32), blkT, 0, stream>>>(W2, w2b, F_, D_);

  // QKV projections (V written transposed for the PV matmul)
  dim3 gDD(D_ / 64, M_ / 128);
  gemm_wmma<0><<<gDD, blk256, 0, stream>>>(xb, wqb, bq, qb,  nullptr, M_, D_, D_);
  gemm_wmma<0><<<gDD, blk256, 0, stream>>>(xb, wkb, bk, kb2, nullptr, M_, D_, D_);
  gemm_wmma<3><<<gDD, blk256, 0, stream>>>(xb, wvb, bv, vt,  nullptr, M_, D_, D_);

  // harmonic flash attention
  dim3 gFA(S_ / 64, B_ * H_);
  flash_attn<<<gFA, blk128, 0, stream>>>(qb, kb2, vt, res, phs, at16);

  // output projection (f32)
  gemm_wmma<1><<<gDD, blk256, 0, stream>>>(at16, wob, bo, nullptr, aprj, M_, D_, D_);

  // LN1
  harmonic_ln<<<M_, blk256, 0, stream>>>(x, aprj, g1, b1, s1, y1, y1b);

  // FFN up + spectral activation (f16), FFN down (f32)
  gemm_wmma<2><<<dim3(F_ / 64, M_ / 128), blk256, 0, stream>>>(
      y1b, w1b, bf1, act, nullptr, M_, F_, D_);
  gemm_wmma<1><<<gDD, blk256, 0, stream>>>(act, w2b, bf2, nullptr, ffn, M_, D_, F_);

  // LN2 -> final output
  harmonic_ln<<<M_, blk256, 0, stream>>>(y1, ffn, g2, b2, s2, (float*)d_out, nullptr);
}
